// DistanceGatedScoringFunction_21938692948578
// MI455X (gfx1250) — compile-verified, matrix operand order checked by reading, not run on hardware
//
#include <hip/hip_runtime.h>
#include <hip/hip_bf16.h>

typedef __attribute__((ext_vector_type(16))) _Float16 v16h;
typedef __attribute__((ext_vector_type(8)))  float    v8f;

#define D_DIM 256
#define H_DIM 256
#define E_NUM 8
#define L_DIM 64
#define CROWS 256

union Frag16 {
  v16h v;
  _Float16 h[16];
  uint4 u[2];
};

// ---------------------------------------------------------------------------
// K1: pack f32 weight W[E, K, N] (row-major) into f16 WMMA B-fragment order.
// Fragment (e, ntile, ktile): 32 lanes x 16 halves, contiguous per lane.
// Element j of lane l -> K = kt*32 + (l>>4)*8 + (v&3)*2 + b + (v>=4 ? 16 : 0),
//                        N = nt*16 + (l&15), with v=j>>1, b=j&1.
// (16-bit A-matrix layout from ISA 7.12.2, mirrored for B.)
// ---------------------------------------------------------------------------
__global__ void pack_b_kernel(const float* __restrict__ W,
                              _Float16* __restrict__ dst,
                              int Earg, int Kdim, int Ncols) {
  int idx = blockIdx.x * blockDim.x + threadIdx.x;
  int NT = Ncols >> 4, KT = Kdim >> 5;
  int total = Earg * NT * KT * 512;
  if (idx >= total) return;
  int j    = idx & 15;
  int lane = (idx >> 4) & 31;
  int frag = idx >> 9;
  int kt = frag % KT;
  int nt = (frag / KT) % NT;
  int e  = frag / (KT * NT);
  int v  = j >> 1, b = j & 1;
  int K  = kt * 32 + ((lane >> 4) * 8) + ((v & 3) * 2 + b) + ((v >= 4) ? 16 : 0);
  int Nc = nt * 16 + (lane & 15);
  dst[idx] = (_Float16)W[((size_t)e * Kdim + K) * Ncols + Nc];
}

// ---------------------------------------------------------------------------
// K2: gating. Per 64-row block: WMMA gate projection (256 -> 64), then
// distance-to-centers, E->H->E MLP, gumbel-softmax. Writes probs[N,8] and
// zeroes the global accumulator.
// ---------------------------------------------------------------------------
__global__ __launch_bounds__(256) void gating_kernel(
    const float* __restrict__ gate_in, const float* __restrict__ gumbel,
    const _Float16* __restrict__ pWg, const float* __restrict__ bg,
    const float* __restrict__ centers,
    const float* __restrict__ Wgm1, const float* __restrict__ bgm1,
    const float* __restrict__ Wgm2, const float* __restrict__ bgm2,
    float* __restrict__ probs, float* __restrict__ acc, int Nrows) {
  __shared__ _Float16 gsm[64][D_DIM];   // 32 KB  gate rows (f16)
  __shared__ float    glm[64][L_DIM];   // 16 KB  gate latent
  __shared__ float    lg2[64][E_NUM];   //  2 KB  MLP partial sums

  int tid = threadIdx.x;
  int row0 = blockIdx.x * 64;
  int rows = Nrows - row0; if (rows > 64) rows = 64;
  int mtiles = rows >> 4;

  for (int i = tid; i < rows * D_DIM; i += 256) {
    int r = i >> 8, c = i & 255;
    gsm[r][c] = (_Float16)gate_in[(size_t)(row0 + r) * D_DIM + c];
  }
  for (int i = tid; i < 64 * E_NUM; i += 256) lg2[i >> 3][i & 7] = 0.f;
  __syncthreads();

  int wave = tid >> 5, lane = tid & 31;
  int lrow = lane & 15, lgrp = lane >> 4;

  // 16 (mt, nt) tiles split over 8 waves, 2 each
  for (int t = wave * 2; t < wave * 2 + 2; ++t) {
    int mt = t >> 2, nt = t & 3;
    if (mt >= mtiles) continue;
    Frag16 a[8];
#pragma unroll
    for (int k = 0; k < 8; ++k) {
      const _Float16* p = &gsm[mt * 16 + lrow][k * 32 + lgrp * 8];
      a[k].u[0] = *(const uint4*)p;          // K run [k0, k0+8)
      a[k].u[1] = *(const uint4*)(p + 16);   // K run [k0+16, k0+24)
    }
    v8f c = {};
#pragma unroll
    for (int k = 0; k < 8; ++k) {
      Frag16 bf;
      const _Float16* p = pWg + ((size_t)(nt * 8 + k)) * 512 + lane * 16;
      bf.u[0] = *(const uint4*)p;
      bf.u[1] = *(const uint4*)(p + 8);
      c = __builtin_amdgcn_wmma_f32_16x16x32_f16(false, a[k].v, false, bf.v,
                                                 (short)0, c, false, false);
    }
    int col = nt * 16 + lrow;
    float bgc = bg[col];
#pragma unroll
    for (int r = 0; r < 8; ++r)
      glm[mt * 16 + r + (lgrp ? 8 : 0)][col] = c[r] + bgc;
  }
  __syncthreads();

  // per-row: -||gl - c_e||^2 -> MLP(E->H->E), 4 threads per row split over H
  int prow = tid & 63;
  int q = tid >> 6;
  if (prow < rows) {
    float lg[E_NUM];
    float ss = 0.f;
    for (int l = 0; l < L_DIM; ++l) { float g = glm[prow][l]; ss += g * g; }
    for (int e = 0; e < E_NUM; ++e) {
      float dot = 0.f, cc = 0.f;
      for (int l = 0; l < L_DIM; ++l) {
        float cv = centers[e * L_DIM + l];
        dot += glm[prow][l] * cv; cc += cv * cv;
      }
      lg[e] = -(ss - 2.f * dot + cc);
    }
    float part[E_NUM];
    for (int e = 0; e < E_NUM; ++e) part[e] = 0.f;
    for (int h = q * 64; h < q * 64 + 64; ++h) {
      float aa = bgm1[h];
      for (int e = 0; e < E_NUM; ++e) aa += lg[e] * Wgm1[e * H_DIM + h];
      aa = fmaxf(aa, 0.f);
      for (int e = 0; e < E_NUM; ++e) part[e] += aa * Wgm2[h * E_NUM + e];
    }
    for (int e = 0; e < E_NUM; ++e) atomicAdd(&lg2[prow][e], part[e]);
  }
  __syncthreads();

  if (tid < rows) {
    int grow = row0 + tid;
    float z[E_NUM], zmax = -1e30f;
    for (int e = 0; e < E_NUM; ++e) {
      z[e] = lg2[tid][e] + bgm2[e] + gumbel[(size_t)grow * E_NUM + e];
      zmax = fmaxf(zmax, z[e]);
    }
    float s = 0.f;
    for (int e = 0; e < E_NUM; ++e) { z[e] = __expf(z[e] - zmax); s += z[e]; }
    float inv = 1.f / s;
    for (int e = 0; e < E_NUM; ++e)
      probs[(size_t)grow * E_NUM + e] = z[e] * inv;
    acc[grow] = 0.f;   // zero pre-sigmoid accumulator
  }
}

// ---------------------------------------------------------------------------
// K3: expert MLPs. grid = (row chunks of 256) x 8 experts, 8 waves/WG.
// Each wave owns 2 of 16 H-column tiles; B fragments live in VGPRs for the
// whole chunk (loaded from L2 once). x tiles are staged cooperatively into
// LDS as f16 (once per 32 rows, not once per wave) to kill the 8x redundant
// global reads. Layer 3 (H->1) folded into the layer-2 epilogue.
// LDS: 128 KB h1 + 16 KB x-stage + 1 KB eo = 145 KB -> 2 WGs / WGP.
// ---------------------------------------------------------------------------
__global__ __launch_bounds__(256) void expert_kernel(
    const float* __restrict__ x,
    const _Float16* __restrict__ pW1, const _Float16* __restrict__ pW2,
    const float* __restrict__ be1, const float* __restrict__ be2,
    const float* __restrict__ We3, const float* __restrict__ be3,
    const float* __restrict__ probs, float* __restrict__ acc, int Nrows) {
  __shared__ _Float16 h1s[CROWS][H_DIM];  // 128 KB layer-1 activations (f16)
  __shared__ _Float16 xs[32][D_DIM];      //  16 KB staged x rows (f16)
  __shared__ float    eo[CROWS];          //   1 KB per-row expert output

  int tid = threadIdx.x;
  int e = blockIdx.y;
  int row0 = blockIdx.x * CROWS;
  int rows = Nrows - row0; if (rows > CROWS) rows = CROWS;
  int mtiles = rows >> 4;

  int wave = tid >> 5, lane = tid & 31;
  int lrow = lane & 15, lgrp = lane >> 4;

  for (int i = tid; i < CROWS; i += 256) eo[i] = 0.f;

  // ---- layer 1: h1 = relu(x @ We1[e] + be1[e]) ----
  Frag16 b1[2][8];
#pragma unroll
  for (int nn = 0; nn < 2; ++nn)
#pragma unroll
    for (int k = 0; k < 8; ++k) {
      size_t frag = ((size_t)(e * 16 + wave * 2 + nn)) * 8 + k;
      const _Float16* p = pW1 + frag * 512 + lane * 16;
      b1[nn][k].u[0] = *(const uint4*)p;
      b1[nn][k].u[1] = *(const uint4*)(p + 8);
    }

  for (int mtg = 0; mtg < mtiles; mtg += 2) {
    int grows = (mtiles - mtg >= 2 ? 2 : 1) * 16;   // rows in this group
    __syncthreads();   // previous xs readers done
    // cooperative stage: grows x 256 f32 -> f16 (float4-vectorized)
    for (int i = tid * 4; i < grows * D_DIM; i += 256 * 4) {
      int r = i >> 8, c = i & 255;
      float4 f = *(const float4*)(x + (size_t)(row0 + mtg * 16 + r) * D_DIM + c);
      xs[r][c + 0] = (_Float16)f.x;
      xs[r][c + 1] = (_Float16)f.y;
      xs[r][c + 2] = (_Float16)f.z;
      xs[r][c + 3] = (_Float16)f.w;
    }
    __syncthreads();
    for (int sub = 0; sub < 2; ++sub) {
      int mt = mtg + sub;
      if (mt >= mtiles) break;
      Frag16 a[8];
#pragma unroll
      for (int k = 0; k < 8; ++k) {
        const _Float16* p = &xs[sub * 16 + lrow][k * 32 + lgrp * 8];
        a[k].u[0] = *(const uint4*)p;
        a[k].u[1] = *(const uint4*)(p + 16);
      }
#pragma unroll
      for (int nn = 0; nn < 2; ++nn) {
        v8f c = {};
#pragma unroll
        for (int k = 0; k < 8; ++k)
          c = __builtin_amdgcn_wmma_f32_16x16x32_f16(false, a[k].v, false,
                                                     b1[nn][k].v, (short)0, c,
                                                     false, false);
        int col = (wave * 2 + nn) * 16 + lrow;
        float bb = be1[e * H_DIM + col];
#pragma unroll
        for (int r = 0; r < 8; ++r)
          h1s[mt * 16 + r + (lgrp ? 8 : 0)][col] =
              (_Float16)fmaxf(c[r] + bb, 0.f);
      }
    }
  }
  __syncthreads();

  // ---- layer 2 + fold: eo = relu(h1 @ We2[e] + be2[e]) @ We3[e] ----
  Frag16 b2[2][8];
#pragma unroll
  for (int nn = 0; nn < 2; ++nn)
#pragma unroll
    for (int k = 0; k < 8; ++k) {
      size_t frag = ((size_t)(e * 16 + wave * 2 + nn)) * 8 + k;
      const _Float16* p = pW2 + frag * 512 + lane * 16;
      b2[nn][k].u[0] = *(const uint4*)p;
      b2[nn][k].u[1] = *(const uint4*)(p + 8);
    }
  float w3[2], bb2[2];
#pragma unroll
  for (int nn = 0; nn < 2; ++nn) {
    int col = (wave * 2 + nn) * 16 + lrow;
    w3[nn]  = We3[e * H_DIM + col];
    bb2[nn] = be2[e * H_DIM + col];
  }

  for (int mt = 0; mt < mtiles; ++mt) {
    Frag16 a[8];
#pragma unroll
    for (int k = 0; k < 8; ++k) {
      const _Float16* p = &h1s[mt * 16 + lrow][k * 32 + lgrp * 8];
      a[k].u[0] = *(const uint4*)p;
      a[k].u[1] = *(const uint4*)(p + 16);
    }
#pragma unroll
    for (int nn = 0; nn < 2; ++nn) {
      v8f c = {};
#pragma unroll
      for (int k = 0; k < 8; ++k)
        c = __builtin_amdgcn_wmma_f32_16x16x32_f16(false, a[k].v, false,
                                                   b2[nn][k].v, (short)0, c,
                                                   false, false);
#pragma unroll
      for (int r = 0; r < 8; ++r) {
        float vv = fmaxf(c[r] + bb2[nn], 0.f) * w3[nn];
        vv += __shfl_xor(vv, 1);
        vv += __shfl_xor(vv, 2);
        vv += __shfl_xor(vv, 4);
        vv += __shfl_xor(vv, 8);               // reduce across 16-lane half
        if (lrow == 0)
          atomicAdd(&eo[mt * 16 + r + (lgrp ? 8 : 0)], vv);
      }
    }
  }
  __syncthreads();

  for (int i = tid; i < rows; i += 256) {
    float val = (eo[i] + be3[e]) * probs[(size_t)(row0 + i) * E_NUM + e];
    atomicAdd(&acc[row0 + i], val);
  }
}

// ---------------------------------------------------------------------------
// K4: out = sigmoid(acc)
// ---------------------------------------------------------------------------
__global__ void sigmoid_kernel(const float* __restrict__ acc,
                               float* __restrict__ out, int Nrows) {
  int i = blockIdx.x * blockDim.x + threadIdx.x;
  if (i < Nrows) out[i] = 1.f / (1.f + __expf(-acc[i]));
}

extern "C" void kernel_launch(void* const* d_in, const int* in_sizes, int n_in,
                              void* d_out, int out_size, void* d_ws, size_t ws_size,
                              hipStream_t stream) {
  const float* score   = (const float*)d_in[0];
  const float* gatein  = (const float*)d_in[1];
  const float* gumbel  = (const float*)d_in[2];
  const float* Wg      = (const float*)d_in[3];
  const float* bg      = (const float*)d_in[4];
  const float* centers = (const float*)d_in[5];
  const float* Wgm1    = (const float*)d_in[6];
  const float* bgm1    = (const float*)d_in[7];
  const float* Wgm2    = (const float*)d_in[8];
  const float* bgm2    = (const float*)d_in[9];
  const float* We1     = (const float*)d_in[10];
  const float* be1     = (const float*)d_in[11];
  const float* We2     = (const float*)d_in[12];
  const float* be2     = (const float*)d_in[13];
  const float* We3     = (const float*)d_in[14];
  const float* be3     = (const float*)d_in[15];
  int N = in_sizes[0] / D_DIM;

  // workspace layout
  const size_t W_HALVES = (size_t)E_NUM * 16 * 8 * 512;   // 524288 per layer
  _Float16* pW1 = (_Float16*)d_ws;
  _Float16* pW2 = pW1 + W_HALVES;
  _Float16* pWg = pW2 + W_HALVES;                         // + 16384 halves
  float* probs  = (float*)(pWg + (size_t)4 * 8 * 512);
  float* acc    = probs + (size_t)N * E_NUM;

  pack_b_kernel<<<(int)((W_HALVES + 255) / 256), 256, 0, stream>>>(
      We1, pW1, E_NUM, D_DIM, H_DIM);
  pack_b_kernel<<<(int)((W_HALVES + 255) / 256), 256, 0, stream>>>(
      We2, pW2, E_NUM, H_DIM, H_DIM);
  pack_b_kernel<<<(4 * 8 * 512 + 255) / 256, 256, 0, stream>>>(
      Wg, pWg, 1, D_DIM, L_DIM);

  gating_kernel<<<(N + 63) / 64, 256, 0, stream>>>(
      gatein, gumbel, pWg, bg, centers, Wgm1, bgm1, Wgm2, bgm2, probs, acc, N);

  dim3 egrid((N + CROWS - 1) / CROWS, E_NUM);
  expert_kernel<<<egrid, 256, 0, stream>>>(
      score, pW1, pW2, be1, be2, We3, be3, probs, acc, N);

  sigmoid_kernel<<<(N + 255) / 256, 256, 0, stream>>>(acc, (float*)d_out, N);
}
